// MultiheadAttention_40913858461791
// MI455X (gfx1250) — matrix operand order repaired and checked by reading, not compiled, not run on hardware
//
#include <hip/hip_runtime.h>

typedef __attribute__((ext_vector_type(16))) _Float16 v16h;
typedef __attribute__((ext_vector_type(8)))  float    v8f;
typedef __attribute__((ext_vector_type(4)))  int      v4i;

#define WMMA_F16(a, b, c) \
  __builtin_amdgcn_wmma_f32_16x16x32_f16(false, (a), false, (b), (short)0, (c), false, false)

// ---------------------------------------------------------------------------
// Async global->LDS copy (CDNA5 GLOBAL_LOAD_ASYNC_TO_LDS_B128, ASYNCcnt),
// guarded so the build falls back to a synchronous copy if the builtin is
// absent on this toolchain. Per hipcc diagnostic, the builtin takes
// (int4 addrspace(1)*, int4 addrspace(3)*, imm offset, imm cpol).
// ---------------------------------------------------------------------------
#if defined(__has_builtin)
#if __has_builtin(__builtin_amdgcn_global_load_async_to_lds_b128)
#define HAS_ASYNC_LDS 1
#endif
#endif

#if defined(HAS_ASYNC_LDS)
typedef __attribute__((address_space(1))) v4i* g_v4i_p;
typedef __attribute__((address_space(3))) v4i* l_v4i_p;
__device__ __forceinline__ void copy16_g2l(const _Float16* g, _Float16* l) {
  __builtin_amdgcn_global_load_async_to_lds_b128((g_v4i_p)g, (l_v4i_p)l, 0, 0);
}
#if __has_builtin(__builtin_amdgcn_s_wait_asynccnt)
#define WAIT_ASYNCCNT(n) __builtin_amdgcn_s_wait_asynccnt(n)
#else
#define WAIT_ASYNCCNT(n) asm volatile("s_wait_asynccnt %0" ::"i"(n))
#endif
#else
__device__ __forceinline__ void copy16_g2l(const _Float16* g, _Float16* l) {
  *(uint4*)l = *(const uint4*)g;
}
#define WAIT_ASYNCCNT(n) ((void)0)
#endif

// ---------------------------------------------------------------------------
// Fragment loaders, per CDNA5 ISA §7.12.2 (16-bit A 16x32, B 32x16 layouts).
// rowptr points at 32 consecutive f16 (one row's K-chunk); 4B-aligned.
// A: lane m (=lane&15); VGPR j holds K pairs {0,2,4,6,16,18,20,22} (+8 if hi).
// B: lane n (=lane&15); VGPR j holds K pairs {0,2,4,6,16,18,20,22} (+8 if hi).
// ---------------------------------------------------------------------------
__device__ __forceinline__ v16h fragA(const _Float16* rowptr, int hi) {
  union { v16h v; unsigned u[8]; } f;
  const unsigned* p = (const unsigned*)rowptr;
  const int b = hi * 4;
#pragma unroll
  for (int j = 0; j < 4; ++j) { f.u[j] = p[b + j]; f.u[4 + j] = p[8 + b + j]; }
  return f.v;
}

__device__ __forceinline__ v16h fragB(const _Float16* rowptr, int hi) {
  union { v16h v; unsigned u[8]; } f;
  const unsigned* p = (const unsigned*)rowptr;
  const int b = hi * 4;
#pragma unroll
  for (int j = 0; j < 4; ++j) { f.u[j] = p[b + j]; f.u[4 + j] = p[8 + b + j]; }
  return f.v;
}

// ---------------------------------------------------------------------------
// One-time conversions (bandwidth-trivial vs 23.3 TB/s)
// ---------------------------------------------------------------------------
__global__ void cvt_f32_f16(const float* __restrict__ in, _Float16* __restrict__ out, int n) {
  for (int i = blockIdx.x * blockDim.x + threadIdx.x; i < n; i += gridDim.x * blockDim.x)
    out[i] = (_Float16)in[i];
}

// in[K][N] f32 -> out[N][K] f16 (32x32 LDS tiles)
__global__ __launch_bounds__(256) void cvt_transpose(
    const float* __restrict__ in, _Float16* __restrict__ out, int K, int N) {
  __shared__ float tile[32][33];
  const int bx = blockIdx.x * 32;  // N
  const int by = blockIdx.y * 32;  // K
  const int tx = threadIdx.x & 31, ty = threadIdx.x >> 5;
#pragma unroll
  for (int i = 0; i < 32; i += 8)
    tile[ty + i][tx] = in[(size_t)(by + ty + i) * N + bx + tx];
  __syncthreads();
#pragma unroll
  for (int i = 0; i < 32; i += 8)
    out[(size_t)(bx + ty + i) * K + by + tx] = (_Float16)tile[tx][ty + i];
}

// V part of qkv16 [b*2048+key][2048 + h*64 + dc] -> vT [(b,h)][dc][key]
__global__ __launch_bounds__(256) void transpose_v(
    const _Float16* __restrict__ qkv, _Float16* __restrict__ vT) {
  __shared__ _Float16 tile[32][33];
  const int bh = blockIdx.z;                  // b*16+h
  const int keyb = blockIdx.x * 32;           // 0..2047
  const int dcb = blockIdx.y * 32;            // 0 or 32
  const int tx = threadIdx.x & 31, ty = threadIdx.x >> 5;
  const _Float16* src = qkv + ((size_t)(bh >> 4) * 2048) * 3072 + 2048 + (bh & 15) * 64;
#pragma unroll
  for (int i = 0; i < 32; i += 8)
    tile[ty + i][tx] = src[(size_t)(keyb + ty + i) * 3072 + dcb + tx];
  __syncthreads();
  _Float16* dst = vT + ((size_t)bh * 64 + dcb) * 2048 + keyb;
#pragma unroll
  for (int i = 0; i < 32; i += 8)
    dst[(size_t)(ty + i) * 2048 + tx] = tile[tx][ty + i];
}

// ---------------------------------------------------------------------------
// Tiled WMMA GEMM: C[M,N] = A[M,K] * BT[N,K]^T; A,BT f16; C f16 or f32(+bias).
// Block tile 128x128, 8 waves (2x4), wave tile 64x32, BK=32 (one WMMA step).
// Both operands staged with natural b128 copies (async + double buffered).
// ---------------------------------------------------------------------------
template <bool OUT_F32>
__global__ __launch_bounds__(256) void gemm_f16(
    const _Float16* __restrict__ A, const _Float16* __restrict__ BT,
    void* __restrict__ Cout, const float* __restrict__ bias, int M, int N, int K) {
  __shared__ _Float16 As[2][128][32];  // [m][k]
  __shared__ _Float16 Bs[2][128][32];  // [n][k]

  const int t = threadIdx.x;
  const int lane = t & 31, wave = t >> 5;
  const int wm = wave >> 2, wn = wave & 3;  // 2 x 4 wave grid
  const int hi = lane >> 4, l16 = lane & 15;
  const int mbase = blockIdx.y * 128;
  const int nbase = blockIdx.x * 128;

  v8f acc[4][2];
  const v8f zero = {};
#pragma unroll
  for (int i = 0; i < 4; ++i) { acc[i][0] = zero; acc[i][1] = zero; }

  // per tile, per thread: 2 x b128 for A + 2 x b128 for B (4 async ops)
  auto stage = [&](int buf, int kb) {
#pragma unroll
    for (int i = 0; i < 2; ++i) {
      const int idx = i * 256 + t;
      const int row = idx >> 2, ch = (idx & 3) * 8;
      copy16_g2l(A + (size_t)(mbase + row) * K + kb + ch, &As[buf][row][ch]);
      copy16_g2l(BT + (size_t)(nbase + row) * K + kb + ch, &Bs[buf][row][ch]);
    }
  };

  const int nk = K / 32;
  stage(0, 0);
  for (int ki = 0; ki < nk; ++ki) {
    const int buf = ki & 1;
    if (ki + 1 < nk) {
      stage(buf ^ 1, (ki + 1) * 32);
      WAIT_ASYNCCNT(4);  // tile ki done; tile ki+1 still in flight
    } else {
      WAIT_ASYNCCNT(0);
    }
    __syncthreads();

    v16h af[4];
#pragma unroll
    for (int ms = 0; ms < 4; ++ms)
      af[ms] = fragA(&As[buf][wm * 64 + ms * 16 + l16][0], hi);
#pragma unroll
    for (int ns = 0; ns < 2; ++ns) {
      const v16h bf = fragB(&Bs[buf][wn * 32 + ns * 16 + l16][0], hi);
#pragma unroll
      for (int ms = 0; ms < 4; ++ms)
        acc[ms][ns] = WMMA_F16(af[ms], bf, acc[ms][ns]);
    }
    __syncthreads();
  }

  // C layout: VGPR r -> row r (lanes 0-15) / r+8 (lanes 16-31); col = lane&15
#pragma unroll
  for (int ms = 0; ms < 4; ++ms)
#pragma unroll
    for (int ns = 0; ns < 2; ++ns)
#pragma unroll
      for (int r = 0; r < 8; ++r) {
        const int row = mbase + wm * 64 + ms * 16 + r + 8 * hi;
        const int col = nbase + wn * 32 + ns * 16 + l16;
        const float v = acc[ms][ns][r];
        if (OUT_F32)
          ((float*)Cout)[(size_t)row * N + col] = v + bias[col];
        else
          ((_Float16*)Cout)[(size_t)row * N + col] = (_Float16)v;
      }
}

// ---------------------------------------------------------------------------
// Flash attention forward. qkv: [B*N, 3072] f16; vT: [(b,h)][64][2048] f16.
// Block = 8 waves handles (b, h, 128 q rows); wave owns 16 q rows.
// K/V tiles (32 keys) async-staged, double buffered; online softmax.
// ---------------------------------------------------------------------------
__global__ __launch_bounds__(256) void attn_fwd(
    const _Float16* __restrict__ qkv, const _Float16* __restrict__ vT,
    _Float16* __restrict__ o16) {
  __shared__ _Float16 Ks[2][32][64];     // [key][dc]
  __shared__ _Float16 Vt[2][64][32];     // [dc][key]
  __shared__ _Float16 Pb[8][16][32];     // per-wave P re-layout scratch

  const int t = threadIdx.x;
  const int lane = t & 31, wave = t >> 5;
  const int hi = lane >> 4, l16 = lane & 15;

  const int qblk = blockIdx.x;  // 16 q-blocks of 128
  const int h = blockIdx.y;     // 16 heads
  const int b = blockIdx.z;     // 4 batches

  const size_t baserow = (size_t)b * 2048;
  const int qrow = qblk * 128 + wave * 16;

  // Q A-fragments (16 rows x 64 K-dim = two chunks), loaded once from global
  const _Float16* qptr = qkv + (baserow + qrow + l16) * 3072 + h * 64;
  const v16h aq0 = fragA(qptr, hi);
  const v16h aq1 = fragA(qptr + 32, hi);

  float mrow[8], lrow[8];
  v8f o[4];
  const v8f vzero = {};
#pragma unroll
  for (int r = 0; r < 8; ++r) { mrow[r] = -__builtin_inff(); lrow[r] = 0.0f; }
#pragma unroll
  for (int d = 0; d < 4; ++d) o[d] = vzero;

  // staging maps: K tile 32x64 halves -> 1 b128/thread; V tile 64x32 -> 1/thread
  const int kkey = t >> 3, kch = (t & 7) * 8;
  const _Float16* kbase = qkv + (baserow + kkey) * 3072 + 1024 + h * 64 + kch;
  const int vdc = t >> 2, vch = (t & 3) * 8;
  const _Float16* vbase = vT + ((size_t)(b * 16 + h) * 64 + vdc) * 2048 + vch;

  auto stage = [&](int buf, int kt) {
    copy16_g2l(kbase + (size_t)kt * 3072, &Ks[buf][kkey][kch]);
    copy16_g2l(vbase + kt, &Vt[buf][vdc][vch]);
  };

  stage(0, 0);
  for (int ti = 0; ti < 64; ++ti) {  // 2048 keys / 32
    const int buf = ti & 1;
    if (ti + 1 < 64) {
      stage(buf ^ 1, (ti + 1) * 32);
      WAIT_ASYNCCNT(2);
    } else {
      WAIT_ASYNCCNT(0);
    }
    __syncthreads();

    // S(16x32) = Q(16x64) . K^T : two 16-key subtiles, K-dim 64 = 2 WMMAs each
    v8f s[2];
#pragma unroll
    for (int nt = 0; nt < 2; ++nt) {
      v8f c = {};
      c = WMMA_F16(aq0, fragB(&Ks[buf][nt * 16 + l16][0], hi), c);
      c = WMMA_F16(aq1, fragB(&Ks[buf][nt * 16 + l16][32], hi), c);
      s[nt] = c;
    }

    // Online softmax (rows live in one VGPR across a 16-lane half-wave)
    float alpha[8];
#pragma unroll
    for (int r = 0; r < 8; ++r) {
      const float s0 = s[0][r] * 0.125f;  // SCALE = 64^-0.5
      const float s1 = s[1][r] * 0.125f;
      float rm = fmaxf(s0, s1);
#pragma unroll
      for (int msk = 1; msk <= 8; msk <<= 1)
        rm = fmaxf(rm, __shfl_xor(rm, msk, 16));
      const float mnew = fmaxf(mrow[r], rm);
      alpha[r] = __expf(mrow[r] - mnew);
      const float p0 = __expf(s0 - mnew);
      const float p1 = __expf(s1 - mnew);
      s[0][r] = p0; s[1][r] = p1;
      float rs = p0 + p1;
#pragma unroll
      for (int msk = 1; msk <= 8; msk <<= 1)
        rs += __shfl_xor(rs, msk, 16);
      lrow[r] = lrow[r] * alpha[r] + rs;
      mrow[r] = mnew;
    }
#pragma unroll
    for (int d = 0; d < 4; ++d)
#pragma unroll
      for (int r = 0; r < 8; ++r) o[d][r] *= alpha[r];

    // Re-layout P: C-fragment -> per-wave LDS -> A-fragment (wave-private)
#pragma unroll
    for (int nt = 0; nt < 2; ++nt)
#pragma unroll
      for (int r = 0; r < 8; ++r)
        Pb[wave][r + 8 * hi][nt * 16 + l16] = (_Float16)s[nt][r];
    asm volatile("s_wait_dscnt 0" ::: "memory");  // same-wave DS RAW ordering
    const v16h pa = fragA(&Pb[wave][l16][0], hi);

    // O += P(16x32) . V(32x64): 4 d-subtiles
#pragma unroll
    for (int d = 0; d < 4; ++d)
      o[d] = WMMA_F16(pa, fragB(&Vt[buf][d * 16 + l16][0], hi), o[d]);
    __syncthreads();
  }

  float inv[8];
#pragma unroll
  for (int r = 0; r < 8; ++r) inv[r] = 1.0f / lrow[r];
#pragma unroll
  for (int d = 0; d < 4; ++d)
#pragma unroll
    for (int r = 0; r < 8; ++r)
      o16[(baserow + qrow + r + 8 * hi) * 1024 + h * 64 + d * 16 + l16] =
          (_Float16)(o[d][r] * inv[r]);
}

// ---------------------------------------------------------------------------
extern "C" void kernel_launch(void* const* d_in, const int* in_sizes, int n_in,
                              void* d_out, int out_size, void* d_ws, size_t ws_size,
                              hipStream_t stream) {
  const float* x     = (const float*)d_in[0];   // [4,2048,1024]
  const float* w_qkv = (const float*)d_in[1];   // [1024,3072]
  const float* w_out = (const float*)d_in[2];   // [1024,1024]
  const float* b_out = (const float*)d_in[3];   // [1024]

  char* ws = (char*)d_ws;
  _Float16* x16 = (_Float16*)ws;    ws += (size_t)8192 * 1024 * 2;   // 16 MB
  _Float16* wqT = (_Float16*)ws;    ws += (size_t)3072 * 1024 * 2;   // 6 MB  [3072][1024]
  _Float16* woT = (_Float16*)ws;    ws += (size_t)1024 * 1024 * 2;   // 2 MB  [1024][1024]
  _Float16* qkv16 = (_Float16*)ws;  ws += (size_t)8192 * 3072 * 2;   // 48 MB
  _Float16* vT = (_Float16*)ws;     ws += (size_t)64 * 64 * 2048 * 2;// 16 MB [(b,h)][64][2048]
  _Float16* o16 = (_Float16*)ws;    // 16 MB                          // ~104 MB total

  cvt_f32_f16<<<4096, 256, 0, stream>>>(x, x16, 8192 * 1024);
  cvt_transpose<<<dim3(96, 32), 256, 0, stream>>>(w_qkv, wqT, 1024, 3072);
  cvt_transpose<<<dim3(32, 32), 256, 0, stream>>>(w_out, woT, 1024, 1024);

  // QKV projection: [8192,1024] x [1024,3072] -> qkv16 (f16)
  gemm_f16<false><<<dim3(24, 64), 256, 0, stream>>>(x16, wqT, qkv16, nullptr,
                                                    8192, 3072, 1024);
  // V -> [(b,h)][dc][key] for contiguous B-fragment staging
  transpose_v<<<dim3(64, 2, 64), 256, 0, stream>>>(qkv16, vT);
  // Flash attention -> o16 [8192,1024] f16
  attn_fwd<<<dim3(16, 16, 4), 256, 0, stream>>>(qkv16, vT, o16);
  // Output projection + bias: [8192,1024] x [1024,1024] -> d_out (f32)
  gemm_f16<true><<<dim3(8, 64), 256, 0, stream>>>(o16, woT, d_out, b_out,
                                                  8192, 1024, 1024);
}
